// TreeAttention_69836168233365
// MI455X (gfx1250) — matrix operand order repaired and checked by reading, not compiled, not run on hardware
//
#include <hip/hip_runtime.h>
#include <hip/hip_bf16.h>
#include <math.h>

typedef __attribute__((ext_vector_type(16))) _Float16 v16h;
typedef __attribute__((ext_vector_type(8)))  float    v8f;

#define NB    8
#define TDST  512
#define TSRC  4096
#define HID   64
#define KSEL  64
#define NEGV  (-32000.0f)
#define FMINV (-1e30f)
#define SENT  0x7FFFFFFF
#define ZMAX  128
#define CANDN 256

#define CVT4(dst, o, f) { (dst)[(o)+0] = (_Float16)(f).x; (dst)[(o)+1] = (_Float16)(f).y; \
                          (dst)[(o)+2] = (_Float16)(f).z; (dst)[(o)+3] = (_Float16)(f).w; }

__device__ __forceinline__ v8f wmma_f16(v16h a, v16h b, v8f c) {
  // D = A(16x32 f16) x B(32x16 f16) + C(16x16 f32)
  return __builtin_amdgcn_wmma_f32_16x16x32_f16(false, a, false, b, (short)0, c, false, false);
}

__global__ __launch_bounds__(32) void tree_attn_kernel(
    const float* __restrict__ q, const float* __restrict__ k,
    const float* __restrict__ v, float* __restrict__ out)
{
  __shared__ float qf[HID];
  __shared__ int   pix[ZMAX];
  __shared__ int   pmk[ZMAX];
  __shared__ float sk[ZMAX];
  __shared__ int   si[ZMAX];
  __shared__ int   cd[CANDN];
  __shared__ int   sums[32];
  __shared__ float pw[KSEL];

  const int lane = threadIdx.x;
  const int blk  = blockIdx.x;
  const int n    = blk >> 9;      // / TDST
  const int a    = blk & 511;     // % TDST
  const float tsrcf = (float)(TSRC - TDST + 1 + a);

  const float* qrow  = q + ((size_t)(n * TDST + a)) * HID;
  const float* kbase = k + (size_t)n * TSRC * HID;
  const float* vbase = v + (size_t)n * TSRC * HID;

  // ---- load q into LDS; init pixels/pmask (iter0: Z=96, first 64 valid) ----
  qf[lane]      = qrow[lane];
  qf[lane + 32] = qrow[lane + 32];
  for (int e = 0; e < 4; ++e) {
    int i = lane + e * 32;
    pix[i] = (i < 96) ? i : 0;
    pmk[i] = (i < 64) ? 1 : 0;
  }
  __syncthreads();

  // ---- build WMMA B operands: q replicated across all 16 columns ----
  // B is 32x16 (KxN): lanes 0-15 hold K=c*32+0..7,16..23, lanes 16-31 hold K=c*32+8..15,24..31
  v16h b0, b1;
  {
    const int gb = (lane & 16) ? 8 : 0;
#pragma unroll
    for (int i2 = 0; i2 < 16; ++i2) {
      b0[i2] = (_Float16)qf[gb + i2 + ((i2 >= 8) ? 8 : 0)];
      b1[i2] = (_Float16)qf[32 + gb + i2 + ((i2 >= 8) ? 8 : 0)];
    }
  }

  // Raw scores for `ntile` tiles of 16 pixels each; writes sk[0..16*ntile).
  auto score_tiles = [&](int ntile, float r) {
    for (int t = 0; t < ntile; ++t) {
      const int m = lane & 15;
      const int z = t * 16 + m;
      int p  = pix[z];
      int tx = (int)rintf((float)p * r);
      tx = (tx > (TSRC - 1)) ? (TSRC - 1) : tx;
      const float* kr = kbase + (size_t)tx * HID;
      // A 16x32 f16 layout: lanes0-15 -> K off..off+7 and off+16..off+23 (off=0);
      // lanes16-31 -> off=8.
      const int off = (lane & 16) ? 8 : 0;
      v16h a0, a1;
      float4 f;
      f = *(const float4*)(kr + off +  0); CVT4(a0,  0, f);
      f = *(const float4*)(kr + off +  4); CVT4(a0,  4, f);
      f = *(const float4*)(kr + off + 16); CVT4(a0,  8, f);
      f = *(const float4*)(kr + off + 20); CVT4(a0, 12, f);
      f = *(const float4*)(kr + off + 32); CVT4(a1,  0, f);
      f = *(const float4*)(kr + off + 36); CVT4(a1,  4, f);
      f = *(const float4*)(kr + off + 48); CVT4(a1,  8, f);
      f = *(const float4*)(kr + off + 52); CVT4(a1, 12, f);
      v8f dacc = {};
      dacc = wmma_f16(a0, b0, dacc);
      dacc = wmma_f16(a1, b1, dacc);
      // D: lanes0-15 N=lane, VGPR r -> M=r; lanes16-31 -> M=8+r. Take column 0.
      if (lane == 0) {
#pragma unroll
        for (int rr = 0; rr < 8; ++rr) sk[t * 16 + rr] = dacc[rr];
      } else if (lane == 16) {
#pragma unroll
        for (int rr = 0; rr < 8; ++rr) sk[t * 16 + 8 + rr] = dacc[rr];
      }
    }
  };

  float ws = 64.0f;
  const int tksmax_tab[7] = {63, 96, 96, 96, 96, 96, 64};
  const int Z_tab[7]      = {96, 128, 128, 128, 128, 128, 128};
  const int maxz_tab[7]   = {128, 128, 128, 128, 128, 128, 64};

  for (int it = 0; it < 7; ++it) {
    const bool is_last = (it == 6);
    const int  Zi      = Z_tab[it];
    const int  tks_max = tksmax_tab[it];
    const int  max_z   = maxz_tab[it];
    const float mult   = is_last ? 1.0f : 1.5f;
    const float r      = tsrcf / ws;

    // 1) scores (WMMA)
    score_tiles(Zi >> 4, r);
    __syncthreads();

    // 2) mask + pad + indices
    for (int e = 0; e < 4; ++e) {
      int i = lane + e * 32;
      float s = (i < Zi) ? (pmk[i] ? sk[i] : NEGV) : -1e38f;
      sk[i] = s;
      si[i] = i;
    }
    __syncthreads();

    // 3) stable descending bitonic sort of 128 (score, idx) pairs
    for (int ksz = 2; ksz <= 128; ksz <<= 1)
      for (int j = ksz >> 1; j > 0; j >>= 1) {
        for (int e = 0; e < 4; ++e) {
          int i  = lane + e * 32;
          int ix = i ^ j;
          if (ix > i) {
            float sa = sk[i], sb2 = sk[ix];
            int   ia = si[i], ib  = si[ix];
            bool before = (sa > sb2) || (sa == sb2 && ia < ib);
            bool up     = ((i & ksz) == 0);
            if (up != before) { sk[i] = sb2; sk[ix] = sa; si[i] = ib; si[ix] = ia; }
          }
        }
        __syncthreads();
      }

    // 4) per-query keep count and scale-up parameters
    float upper = fminf(ws - 1.0f, (float)(Zi - 1));
    float tkf   = rintf(ws / tsrcf * 64.0f * mult);
    tkf = fmaxf(1.0f, fminf(tkf, upper));
    const int   tks   = (int)tkf;
    const float wsn   = fminf(tsrcf, ws * 2.0f);
    const float scale = wsn / ws;

    // 5) candidate children (2 per slot); slots >= Zi get sentinels
    for (int e = 0; e < 4; ++e) {
      int i = lane + e * 32;
      int c0v, c1v;
      if (i < Zi) {
        int pval  = (i < tks_max) ? pix[si[i]] : 0;
        float psf = rintf((float)pval * scale);
        float pef = rintf((float)(pval + 1) * scale);
        int cnt   = (int)(pef - psf);
        int c0    = (int)psf;
        int rep   = (i < tks) ? (((cnt - 1) > 0) ? 1 : 0) : 0;
        c0v = c0; c1v = c0 + rep;
      } else { c0v = SENT; c1v = SENT; }
      cd[2 * i] = c0v; cd[2 * i + 1] = c1v;
    }
    __syncthreads();

    // 6) ascending bitonic sort of 256 ints
    for (int ksz = 2; ksz <= 256; ksz <<= 1)
      for (int j = ksz >> 1; j > 0; j >>= 1) {
        for (int e = 0; e < 8; ++e) {
          int i  = lane + e * 32;
          int ix = i ^ j;
          if (ix > i) {
            int va = cd[i], vb = cd[ix];
            bool before = (va <= vb);
            bool up     = ((i & ksz) == 0);
            if (up != before) { cd[i] = vb; cd[ix] = va; }
          }
        }
        __syncthreads();
      }

    // 7) zero pix/pm, then dedup (flag + scan) and scatter first max_z uniques
    for (int e = 0; e < 4; ++e) { int i = lane + e * 32; pix[i] = 0; pmk[i] = 0; }
    __syncthreads();
    {
      const int base8 = lane * 8;
      int nf[8]; int cnt = 0;
#pragma unroll
      for (int t2 = 0; t2 < 8; ++t2) {
        int i = base8 + t2;
        int f2 = (i == 0) ? 1 : (cd[i] != cd[i - 1]);
        nf[t2] = f2; cnt += f2;
      }
      sums[lane] = cnt;
      __syncthreads();
      for (int off = 1; off < 32; off <<= 1) {
        int add = (lane >= off) ? sums[lane - off] : 0;
        __syncthreads();
        sums[lane] += add;
        __syncthreads();
      }
      int rank = (lane == 0) ? 0 : sums[lane - 1];
#pragma unroll
      for (int t2 = 0; t2 < 8; ++t2) {
        int i = base8 + t2;
        if (nf[t2]) {
          int val = cd[i];
          if (val != SENT && rank < max_z) { pix[rank] = val; pmk[rank] = 1; }
          rank++;
        }
      }
    }
    __syncthreads();
    ws = wsn;
  }

  // ---- final attention over the selected 64 pixels ----
  {
    // prefetch gathered V rows into cache (gfx1250 global_prefetch)
    __builtin_prefetch(vbase + (size_t)pix[lane] * HID, 0, 3);
    __builtin_prefetch(vbase + (size_t)pix[lane + 32] * HID, 0, 3);

    const float r = tsrcf / ws;   // == 1.0 -> gather pixels directly
    score_tiles(4, r);
    __syncthreads();

    float s0 = pmk[lane]      ? sk[lane]      : FMINV;
    float s1 = pmk[lane + 32] ? sk[lane + 32] : FMINV;
    float mx = fmaxf(s0, s1);
    for (int off = 16; off > 0; off >>= 1) mx = fmaxf(mx, __shfl_xor(mx, off, 32));
    float e0 = expf(s0 - mx);
    float e1 = expf(s1 - mx);
    float sm = e0 + e1;
    for (int off = 16; off > 0; off >>= 1) sm += __shfl_xor(sm, off, 32);
    pw[lane]      = e0 / sm;
    pw[lane + 32] = e1 / sm;
    __syncthreads();

    float2 acc = make_float2(0.0f, 0.0f);
    const int h = 2 * lane;
    for (int z = 0; z < KSEL; ++z) {
      float wz = pw[z];
      const float2 vv = *(const float2*)(vbase + (size_t)pix[z] * HID + h);
      acc.x += wz * vv.x;
      acc.y += wz * vv.y;
    }
    float* orow = out + ((size_t)(n * TDST + a)) * HID;
    orow[h]     = acc.x;
    orow[h + 1] = acc.y;
  }
}

extern "C" void kernel_launch(void* const* d_in, const int* in_sizes, int n_in,
                              void* d_out, int out_size, void* d_ws, size_t ws_size,
                              hipStream_t stream) {
  (void)in_sizes; (void)n_in; (void)out_size; (void)d_ws; (void)ws_size;
  const float* q = (const float*)d_in[0];
  const float* k = (const float*)d_in[1];
  const float* v = (const float*)d_in[2];
  float* out = (float*)d_out;
  dim3 grid(NB * TDST);
  dim3 block(32);
  tree_attn_kernel<<<grid, block, 0, stream>>>(q, k, v, out);
}
